// BronxLayer_19877108646320
// MI455X (gfx1250) — compile-verified, matrix operand order checked by reading, not run on hardware
//
#include <hip/hip_runtime.h>
#include <hip/hip_fp16.h>
#include <math.h>

#define NN   2048
#define FD   512
#define HH   8
#define HD   64
#define DXF  256
#define KPAD 544   // 512 LN features + 4 stats + 28 zero pad -> multiple of 32

typedef __attribute__((ext_vector_type(16))) _Float16 v16h;
typedef __attribute__((ext_vector_type(8)))  _Float16 v8h_t;
typedef __attribute__((ext_vector_type(8)))  float    v8f;

// ---------------------------------------------------------------------------
// WMMA helpers (gfx1250, wave32): D = A(16x32 f16) * B(32x16 f16) + C(16x16 f32)
// ---------------------------------------------------------------------------
static __device__ __forceinline__ v8f wmma16(v16h a, v16h b, v8f c) {
  return __builtin_amdgcn_wmma_f32_16x16x32_f16(false, a, false, b,
                                                (short)0, c, false, false);
}

// Per-lane fragment load matching the ISA 16-bit 16x32 A layout (B mirrored:
// lane = column, column n of B == row n of the source operand).
// lane<16 : row/col = lane,    K = {k0+0..7,  k0+16..23}
// lane>=16: row/col = lane-16, K = {k0+8..15, k0+24..31}
static __device__ __forceinline__ v16h ld_frag(const _Float16* p0, int ld,
                                               int rc, int k0, int lane) {
  const int sub = (lane & 16) ? 8 : 0;
  const _Float16* p = p0 + (size_t)rc * ld + k0 + sub;
  v16h f;
  *(v8h_t*)&f       = *(const v8h_t*)(p);        // 16B contiguous
  *((v8h_t*)&f + 1) = *(const v8h_t*)(p + 16);   // 16B contiguous
  return f;
}

static __device__ __forceinline__ float wsum(float v) {
#pragma unroll
  for (int o = 16; o > 0; o >>= 1) v += __shfl_xor(v, o, 32);
  return v;
}

static __device__ __forceinline__ void atomicMaxF(float* a, float v) {
  int* ai = (int*)a;
  int cur = __float_as_int(*a);
  while (__int_as_float(cur) < v) {
    int assumed = cur;
    cur = atomicCAS(ai, assumed, __float_as_int(v));
    if (cur == assumed) break;
  }
}

// ---------------------------------------------------------------------------
// Prologue: dtype conversions
// ---------------------------------------------------------------------------
__global__ __launch_bounds__(256) void cvt16_kernel(const float* __restrict__ s,
                                                    _Float16* __restrict__ d, int n) {
  int i = blockIdx.x * 256 + threadIdx.x;
  if (i < n) d[i] = (_Float16)s[i];
}

__global__ __launch_bounds__(256) void cvt_fc_kernel(const float* __restrict__ Wfc,
                                                     _Float16* __restrict__ WfcH) {
  int i = blockIdx.x * 256 + threadIdx.x;
  if (i < FD * KPAD) {
    int r = i / KPAD, c = i % KPAD;
    WfcH[i] = (_Float16)(c < (FD + 4) ? Wfc[r * (FD + 4) + c] : 0.f);
  }
}

// LayerNorm(h) -> f16 rows (feeds projections only; residual uses raw h)
__global__ __launch_bounds__(256) void ln_h_kernel(const float* __restrict__ h,
                                                   const float* __restrict__ g,
                                                   const float* __restrict__ bta,
                                                   _Float16* __restrict__ hnH) {
  int row  = blockIdx.x * 8 + (threadIdx.x >> 5);
  int lane = threadIdx.x & 31;
  const float* hr = h + (size_t)row * FD;
  float vals[16], s = 0.f;
#pragma unroll
  for (int j = 0; j < 16; ++j) { vals[j] = hr[lane + 32 * j]; s += vals[j]; }
  float mu = wsum(s) * (1.f / FD);
  float q = 0.f;
#pragma unroll
  for (int j = 0; j < 16; ++j) { float d = vals[j] - mu; q += d * d; }
  float rs = rsqrtf(wsum(q) * (1.f / FD) + 1e-5f);
#pragma unroll
  for (int j = 0; j < 16; ++j) {
    int c = lane + 32 * j;
    hnH[(size_t)row * FD + c] = (_Float16)((vals[j] - mu) * rs * g[c] + bta[c]);
  }
}

// L1 normalize x -> xnH [N,256] f16 and transposed xnT [256,N] f16
__global__ __launch_bounds__(256) void l1_x_kernel(const float* __restrict__ x,
                                                   _Float16* __restrict__ xnH,
                                                   _Float16* __restrict__ xnT) {
  int row  = blockIdx.x * 8 + (threadIdx.x >> 5);
  int lane = threadIdx.x & 31;
  const float* xr = x + (size_t)row * DXF;
  float vals[8], s = 0.f;
#pragma unroll
  for (int j = 0; j < 8; ++j) { vals[j] = xr[lane + 32 * j]; s += fabsf(vals[j]); }
  float inv = 1.f / fmaxf(wsum(s), 1e-12f);
#pragma unroll
  for (int j = 0; j < 8; ++j) {
    int c = lane + 32 * j;
    _Float16 v = (_Float16)(vals[j] * inv);
    xnH[(size_t)row * DXF + c] = v;
    xnT[(size_t)c * NN + row]  = v;
  }
}

// ---------------------------------------------------------------------------
// Projections: K/Q -> per-head [8][N][64], V -> transposed [8][64][N]
// One wave per 16x16 output tile; K-loop fully unrolled (16 WMMAs).
// ---------------------------------------------------------------------------
__global__ __launch_bounds__(256) void proj_kernel(
    const _Float16* __restrict__ hnH,
    const _Float16* __restrict__ WkH, const float* __restrict__ bk,
    const _Float16* __restrict__ WqH, const float* __restrict__ bq,
    const _Float16* __restrict__ WvH, const float* __restrict__ bv,
    _Float16* __restrict__ kh, _Float16* __restrict__ qh,
    _Float16* __restrict__ vhT) {
  const int lane = threadIdx.x & 31;
  const int w    = blockIdx.x * 8 + (threadIdx.x >> 5);
  const int m0   = (w >> 5) << 4;          // 128 row tiles
  const int n0   = (w & 31) << 4;          // 32 col tiles
  const int mat  = blockIdx.y;
  const _Float16* W   = mat == 0 ? WkH : (mat == 1 ? WqH : WvH);
  const float*   bias = mat == 0 ? bk  : (mat == 1 ? bq  : bv);

  const int ar = m0 + (lane & 15), br = n0 + (lane & 15);
  v8f acc0 = {}, acc1 = {};
#pragma unroll
  for (int kb = 0; kb < FD; kb += 64) {
    acc0 = wmma16(ld_frag(hnH, FD, ar, kb, lane),      ld_frag(W, FD, br, kb, lane),      acc0);
    acc1 = wmma16(ld_frag(hnH, FD, ar, kb + 32, lane), ld_frag(W, FD, br, kb + 32, lane), acc1);
  }
  v8f acc = acc0 + acc1;

  const int n  = n0 + (lane & 15);
  const float bb = bias[n];
  const int mb = m0 + ((lane >> 4) << 3);
  const int hb = n & 7, y = n >> 3;        // flat col c = y*8 + head
#pragma unroll
  for (int r = 0; r < 8; ++r) {
    _Float16 val = (_Float16)(acc[r] + bb);
    int m = mb + r;
    if      (mat == 0) kh[((size_t)hb * NN + m) * HD + y] = val;
    else if (mat == 1) qh[((size_t)hb * NN + m) * HD + y] = val;
    else               vhT[((size_t)hb * HD + y) * NN + m] = val;
  }
}

// ---------------------------------------------------------------------------
// Fused attention: logits -> head-mix -> head-softmax -> hm/xm/stats.
// Block = 256 threads (8 waves), owns 16 x-rows; wave b owns head b.
// Streams z in 16-tiles; a_h / a_x never hit global memory.
// ---------------------------------------------------------------------------
__global__ __launch_bounds__(256) void attn_fused_kernel(
    const _Float16* __restrict__ kh, const _Float16* __restrict__ qh,
    const _Float16* __restrict__ vhT, const _Float16* __restrict__ xnH,
    const _Float16* __restrict__ xnT,
    const float* __restrict__ Wm1, const float* __restrict__ bm1,
    const float* __restrict__ Wm2, const float* __restrict__ bm2,
    float* __restrict__ hm_raw, float* __restrict__ xm_raw,
    float* __restrict__ stats) {
  __shared__ float    ah_lds[HH][16][16];     // per-head logit tile
  __shared__ float    ax_lds[16][16];         // a_x tile (atomically reduced)
  __shared__ _Float16 p_lds[HH][16][32];      // softmax probs, 2 z-tiles deep
  __shared__ _Float16 axp_lds[16][32];        // mixed a_x, 2 z-tiles deep
  __shared__ float    stat_lds[16][5];        // sum, sum|.|, sumsq, max, diag
  __shared__ float    mix_lds[89];            // Wm1(72) bm1(8) Wm2(8) bm2(1)

  const int lane = threadIdx.x & 31;
  const int wave = threadIdx.x >> 5;          // head id
  const int x0   = blockIdx.x * 16;

  if (threadIdx.x < 72)      mix_lds[threadIdx.x] = Wm1[threadIdx.x];
  else if (threadIdx.x < 80) mix_lds[threadIdx.x] = bm1[threadIdx.x - 72];
  else if (threadIdx.x < 88) mix_lds[threadIdx.x] = Wm2[threadIdx.x - 80];
  else if (threadIdx.x == 88) mix_lds[88] = bm2[0];
  if (threadIdx.x < 16) {
    stat_lds[threadIdx.x][0] = 0.f; stat_lds[threadIdx.x][1] = 0.f;
    stat_lds[threadIdx.x][2] = 0.f; stat_lds[threadIdx.x][3] = -INFINITY;
    stat_lds[threadIdx.x][4] = 0.f;
  }

  // z-invariant A fragments: this head's k rows, this wave's xn K-chunk
  const _Float16* khB = kh + (size_t)wave * NN * HD;
  const _Float16* qhB = qh + (size_t)wave * NN * HD;
  const _Float16* vhB = vhT + (size_t)wave * HD * NN;
  const v16h kA0 = ld_frag(khB, HD, x0 + (lane & 15), 0, lane);
  const v16h kA1 = ld_frag(khB, HD, x0 + (lane & 15), 32, lane);
  const v16h xA  = ld_frag(xnH, DXF, x0 + (lane & 15), wave * 32, lane);

  v8f hmacc[4] = {};   // 16 x-rows x 64 head-dim
  v8f xmacc[2] = {};   // 16 x-rows x 32 xn-cols (this wave's slice)

  __syncthreads();

  for (int zt = 0; zt < NN / 16; ++zt) {
    const int z0  = zt * 16;
    const int par = zt & 1;

    ((float*)ax_lds)[threadIdx.x] = 0.f;
    if (z0 + 16 < NN) {                      // L2 prefetch of next tiles
      __builtin_prefetch(qhB + (size_t)(z0 + 16) * HD, 0, 1);
      __builtin_prefetch(xnH + (size_t)(z0 + 16) * DXF, 0, 1);
    }
    __syncthreads();

    {   // a_h tile for this head: K=64 contraction -> 2 WMMAs
      v8f c = {};
      c = wmma16(kA0, ld_frag(qhB, HD, z0 + (lane & 15), 0, lane), c);
      c = wmma16(kA1, ld_frag(qhB, HD, z0 + (lane & 15), 32, lane), c);
      const int n = lane & 15, mb = (lane >> 4) << 3;
#pragma unroll
      for (int r = 0; r < 8; ++r) ah_lds[wave][mb + r][n] = c[r];
    }
    {   // a_x partial: this wave contracts xn K-chunk [wave*32, wave*32+32)
      v8f c = {};
      c = wmma16(xA, ld_frag(xnH, DXF, z0 + (lane & 15), wave * 32, lane), c);
      const int n = lane & 15, mb = (lane >> 4) << 3;
#pragma unroll
      for (int r = 0; r < 8; ++r) atomicAdd(&ax_lds[mb + r][n], c[r]);
    }
    __syncthreads();

    {   // mixing (9->8) + head softmax + a_x update + stats; 1 thread/(x,z)
      const int m = threadIdx.x >> 4, n = threadIdx.x & 15;
      float ax = ax_lds[m][n];
      float ahv[8];
#pragma unroll
      for (int b = 0; b < 8; ++b) ahv[b] = ah_lds[b][m][n];
      float mixed[8], mx = -INFINITY;
#pragma unroll
      for (int b = 0; b < 8; ++b) {
        float a = mix_lds[72 + b] + mix_lds[b * 9] * ax;
#pragma unroll
        for (int j = 0; j < 8; ++j) a += mix_lds[b * 9 + 1 + j] * ahv[j];
        mixed[b] = a; mx = fmaxf(mx, a);
      }
      float ax2 = mix_lds[88] + ax;
#pragma unroll
      for (int j = 0; j < 8; ++j) ax2 += mix_lds[80 + j] * mixed[j];
      float e[8], es = 0.f;
#pragma unroll
      for (int b = 0; b < 8; ++b) { e[b] = __expf(mixed[b] - mx); es += e[b]; }
      float inv = 1.f / es;
#pragma unroll
      for (int b = 0; b < 8; ++b)
        p_lds[b][m][par * 16 + n] = (_Float16)(e[b] * inv);
      axp_lds[m][par * 16 + n] = (_Float16)ax2;
      atomicAdd(&stat_lds[m][0], ax2);
      atomicAdd(&stat_lds[m][1], fabsf(ax2));
      atomicAdd(&stat_lds[m][2], ax2 * ax2);
      atomicMaxF(&stat_lds[m][3], ax2);
      if (x0 + m == z0 + n) stat_lds[m][4] = ax2;   // single writer
    }
    __syncthreads();

    if (par) {  // consume two staged z-tiles: K=32 contraction over z
      const int zd = z0 - 16;
      const v16h pA = ld_frag(&p_lds[wave][0][0], 32, lane & 15, 0, lane);
      const v16h aA = ld_frag(&axp_lds[0][0], 32, lane & 15, 0, lane);
#pragma unroll
      for (int t = 0; t < 4; ++t)   // hm: 64 head-dim cols (vhT rows contiguous in z)
        hmacc[t] = wmma16(pA, ld_frag(vhB, NN, t * 16 + (lane & 15), zd, lane), hmacc[t]);
#pragma unroll
      for (int t = 0; t < 2; ++t)   // xm: this wave's 32 of 256 cols (xnT)
        xmacc[t] = wmma16(aA,
                          ld_frag(xnT, NN, wave * 32 + t * 16 + (lane & 15), zd, lane),
                          xmacc[t]);
    }
  }
  __syncthreads();

  // ---- epilogue stores -----------------------------------------------------
  const int nlo = lane & 15, mb = x0 + ((lane >> 4) << 3);
#pragma unroll
  for (int t = 0; t < 4; ++t) {
    const int zp = t * 16 + nlo;           // head-dim index; flat col = zp*8 + head
#pragma unroll
    for (int r = 0; r < 8; ++r)
      hm_raw[(size_t)(mb + r) * FD + zp * 8 + wave] = hmacc[t][r];
  }
#pragma unroll
  for (int t = 0; t < 2; ++t) {
    const int c = wave * 32 + t * 16 + nlo;
#pragma unroll
    for (int r = 0; r < 8; ++r)
      xm_raw[(size_t)(mb + r) * DXF + c] = xmacc[t][r];
  }
  if (threadIdx.x < 16) {
#pragma unroll
    for (int s = 0; s < 5; ++s)
      stats[(size_t)(x0 + threadIdx.x) * 5 + s] = stat_lds[threadIdx.x][s];
  }
}

// ---------------------------------------------------------------------------
// xm epilogue: deferred L1 row-scale + residual; compute per-row stats vector i
// ---------------------------------------------------------------------------
__global__ __launch_bounds__(256) void xm_epilogue_kernel(
    const float* __restrict__ xm_raw, const float* __restrict__ stats,
    const float* __restrict__ x, float* __restrict__ xm_out,
    float* __restrict__ iBuf) {
  int row  = blockIdx.x * 8 + (threadIdx.x >> 5);
  int lane = threadIdx.x & 31;
  const float* st = stats + (size_t)row * 5;
  float inv = 1.f / fmaxf(st[1], 1e-12f);
#pragma unroll
  for (int j = 0; j < 8; ++j) {
    int c = lane + 32 * j;
    xm_out[(size_t)row * DXF + c] =
        xm_raw[(size_t)row * DXF + c] * inv + x[(size_t)row * DXF + c];
  }
  if (lane == 0) {
    float sum = st[0], sumsq = st[2], mxv = st[3], dg = st[4];
    float var = (sumsq - sum * sum * (1.f / NN)) * (1.f / (NN - 1));
    iBuf[(size_t)row * 4 + 0] = dg * inv;
    iBuf[(size_t)row * 4 + 1] = sum * inv;
    iBuf[(size_t)row * 4 + 2] = sqrtf(fmaxf(var, 0.f)) * inv;
    iBuf[(size_t)row * 4 + 3] = mxv * inv;
  }
}

// hm residual + LayerNorm + concat(i) + zero pad -> f16 [N,544]; keep hmr f32
__global__ __launch_bounds__(256) void hm_ln_cat_kernel(
    const float* __restrict__ hm_raw, const float* __restrict__ h,
    const float* __restrict__ g, const float* __restrict__ bta,
    const float* __restrict__ iBuf, float* __restrict__ hmr,
    _Float16* __restrict__ hfcat) {
  int row  = blockIdx.x * 8 + (threadIdx.x >> 5);
  int lane = threadIdx.x & 31;
  float vals[16], s = 0.f;
#pragma unroll
  for (int j = 0; j < 16; ++j) {
    int c = lane + 32 * j;
    vals[j] = hm_raw[(size_t)row * FD + c] + h[(size_t)row * FD + c];
    hmr[(size_t)row * FD + c] = vals[j];
    s += vals[j];
  }
  float mu = wsum(s) * (1.f / FD);
  float q = 0.f;
#pragma unroll
  for (int j = 0; j < 16; ++j) { float d = vals[j] - mu; q += d * d; }
  float rs = rsqrtf(wsum(q) * (1.f / FD) + 1e-5f);
#pragma unroll
  for (int j = 0; j < 16; ++j) {
    int c = lane + 32 * j;
    hfcat[(size_t)row * KPAD + c] = (_Float16)((vals[j] - mu) * rs * g[c] + bta[c]);
  }
  hfcat[(size_t)row * KPAD + FD + lane] =
      (lane < 4) ? (_Float16)iBuf[(size_t)row * 4 + lane] : (_Float16)0.f;
}

// FC: [N,544] x [512,544]^T + bias, ELU, + hmr residual -> d_out hf
__global__ __launch_bounds__(256) void fc_kernel(
    const _Float16* __restrict__ hfcat, const _Float16* __restrict__ WfcH,
    const float* __restrict__ bfc, const float* __restrict__ hmr,
    float* __restrict__ out_hf) {
  const int lane = threadIdx.x & 31;
  const int w    = blockIdx.x * 8 + (threadIdx.x >> 5);
  const int m0   = (w >> 5) << 4;
  const int n0   = (w & 31) << 4;
  const int ar = m0 + (lane & 15), br = n0 + (lane & 15);
  v8f acc0 = {}, acc1 = {};
#pragma unroll
  for (int kb = 0; kb < KPAD; kb += 64) {
    acc0 = wmma16(ld_frag(hfcat, KPAD, ar, kb, lane), ld_frag(WfcH, KPAD, br, kb, lane), acc0);
    if (kb + 32 < KPAD)
      acc1 = wmma16(ld_frag(hfcat, KPAD, ar, kb + 32, lane),
                    ld_frag(WfcH, KPAD, br, kb + 32, lane), acc1);
  }
  v8f acc = acc0 + acc1;
  const int n  = n0 + (lane & 15);
  const float bb = bfc[n];
  const int mb = m0 + ((lane >> 4) << 3);
#pragma unroll
  for (int r = 0; r < 8; ++r) {
    float v = acc[r] + bb;
    v = v > 0.f ? v : (__expf(v) - 1.f);               // ELU(alpha=1)
    out_hf[(size_t)(mb + r) * FD + n] = v + hmr[(size_t)(mb + r) * FD + n];
  }
}

// ---------------------------------------------------------------------------
extern "C" void kernel_launch(void* const* d_in, const int* in_sizes, int n_in,
                              void* d_out, int out_size, void* d_ws, size_t ws_size,
                              hipStream_t stream) {
  const float* h    = (const float*)d_in[0];
  const float* x    = (const float*)d_in[1];
  const float* ln_g = (const float*)d_in[2];
  const float* ln_b = (const float*)d_in[3];
  const float* Wk   = (const float*)d_in[4];
  const float* bk   = (const float*)d_in[5];
  const float* Wq   = (const float*)d_in[6];
  const float* bq   = (const float*)d_in[7];
  const float* Wv   = (const float*)d_in[8];
  const float* bv   = (const float*)d_in[9];
  const float* Wm1  = (const float*)d_in[10];
  const float* bm1  = (const float*)d_in[11];
  const float* Wm2  = (const float*)d_in[12];
  const float* bm2  = (const float*)d_in[13];
  const float* lnf_g = (const float*)d_in[14];
  const float* lnf_b = (const float*)d_in[15];
  const float* Wfc  = (const float*)d_in[16];
  const float* bfc  = (const float*)d_in[17];

  float* d_hf = (float*)d_out;                 // [N,512]
  float* d_xm = d_hf + (size_t)NN * FD;        // [N,256]

  size_t off = 0;
  auto alloc = [&](size_t bytes) -> void* {
    void* p = (char*)d_ws + off;
    off += (bytes + 255) & ~(size_t)255;
    return p;
  };
  _Float16* hnH  = (_Float16*)alloc((size_t)NN * FD * 2);
  _Float16* xnH  = (_Float16*)alloc((size_t)NN * DXF * 2);
  _Float16* xnT  = (_Float16*)alloc((size_t)DXF * NN * 2);
  _Float16* WkH  = (_Float16*)alloc((size_t)FD * FD * 2);
  _Float16* WqH  = (_Float16*)alloc((size_t)FD * FD * 2);
  _Float16* WvH  = (_Float16*)alloc((size_t)FD * FD * 2);
  _Float16* WfcH = (_Float16*)alloc((size_t)FD * KPAD * 2);
  _Float16* kh   = (_Float16*)alloc((size_t)HH * NN * HD * 2);
  _Float16* qh   = (_Float16*)alloc((size_t)HH * NN * HD * 2);
  _Float16* vhT  = (_Float16*)alloc((size_t)HH * HD * NN * 2);
  float* hm_raw  = (float*)alloc((size_t)NN * FD * 4);
  float* xm_raw  = (float*)alloc((size_t)NN * DXF * 4);
  float* stats   = (float*)alloc((size_t)NN * 5 * 4);
  float* iBuf    = (float*)alloc((size_t)NN * 4 * 4);
  float* hmr     = (float*)alloc((size_t)NN * FD * 4);
  _Float16* hfcat = (_Float16*)alloc((size_t)NN * KPAD * 2);
  (void)ws_size; (void)n_in; (void)in_sizes; (void)out_size;

  cvt16_kernel<<<(FD * FD + 255) / 256, 256, 0, stream>>>(Wk, WkH, FD * FD);
  cvt16_kernel<<<(FD * FD + 255) / 256, 256, 0, stream>>>(Wq, WqH, FD * FD);
  cvt16_kernel<<<(FD * FD + 255) / 256, 256, 0, stream>>>(Wv, WvH, FD * FD);
  cvt_fc_kernel<<<(FD * KPAD + 255) / 256, 256, 0, stream>>>(Wfc, WfcH);

  ln_h_kernel<<<NN / 8, 256, 0, stream>>>(h, ln_g, ln_b, hnH);
  l1_x_kernel<<<NN / 8, 256, 0, stream>>>(x, xnH, xnT);

  proj_kernel<<<dim3(512, 3), 256, 0, stream>>>(hnH, WkH, bk, WqH, bq, WvH, bv,
                                                kh, qh, vhT);

  attn_fused_kernel<<<NN / 16, 256, 0, stream>>>(kh, qh, vhT, xnH, xnT,
                                                 Wm1, bm1, Wm2, bm2,
                                                 hm_raw, xm_raw, stats);

  xm_epilogue_kernel<<<NN / 8, 256, 0, stream>>>(xm_raw, stats, x, d_xm, iBuf);
  hm_ln_cat_kernel<<<NN / 8, 256, 0, stream>>>(hm_raw, h, lnf_g, lnf_b, iBuf,
                                               hmr, hfcat);
  fc_kernel<<<512, 256, 0, stream>>>(hfcat, WfcH, bfc, hmr, d_hf);
}
